// AttnBlock_1176821039435
// MI455X (gfx1250) — compile-verified, matrix operand order checked by reading, not run on hardware
//
#include <hip/hip_runtime.h>

// ---------------------------------------------------------------------------
// Types for CDNA5 WMMA (wave32): v16bf A/B fragments, v8f C/D accumulators
// ---------------------------------------------------------------------------
typedef __attribute__((ext_vector_type(16))) __bf16 v16bf;
typedef __attribute__((ext_vector_type(8)))  __bf16 v8bf;
typedef __attribute__((ext_vector_type(8)))  float  v8f;

#define C_DIM 512
#define HW    4096
#define BATCH 4
#define M_TOT (BATCH * HW)   // 16384

__device__ __forceinline__ unsigned short f2bf(float f) {
    unsigned u = __builtin_bit_cast(unsigned, f);
    unsigned r = (u + 0x7FFFu + ((u >> 16) & 1u)) >> 16;
    return (unsigned short)r;
}

__device__ __forceinline__ v8f vzero8() {
    v8f r;
#pragma unroll
    for (int i = 0; i < 8; ++i) r[i] = 0.0f;
    return r;
}

// CDNA5 async copy: 16 B per lane, global memory -> LDS, tracked by ASYNCcnt.
// vdst = 32-bit LDS byte offset, vaddr = 64-bit global address.
__device__ __forceinline__ void async_ld16(unsigned lds_off, const void* gaddr) {
    asm volatile("global_load_async_to_lds_b128 %0, %1, off"
                 :: "v"(lds_off), "v"((unsigned long long)(size_t)gaddr)
                 : "memory");
}
#define WAIT_ASYNC(n) asm volatile("s_wait_asynccnt " #n ::: "memory")

// Load a 16x32 bf16 A-style fragment from row-major [rows, ld] bf16 storage.
// ISA 16-bit A 16x32 layout: lanes 0-15 hold K=+0..7 (elems 0-7) and K=+16..23
// (elems 8-15); lanes 16-31 hold K=+8..15 and K=+24..31. Works for B fragments
// too when the source is stored as [N, K] (lane n = column n of B).
__device__ __forceinline__ v16bf ldfrag(const unsigned short* __restrict__ base,
                                        int rowBase, int ld, int kBase) {
    int lane = threadIdx.x & 31;
    const unsigned short* p =
        base + (size_t)(rowBase + (lane & 15)) * ld + kBase + ((lane >> 4) << 3);
    v8bf lo = *(const v8bf*)(p);
    v8bf hi = *(const v8bf*)(p + 16);
    return __builtin_shufflevector(lo, hi, 0, 1, 2, 3, 4, 5, 6, 7,
                                           8, 9, 10, 11, 12, 13, 14, 15);
}

// ---------------------------------------------------------------------------
// Kernel: fp32 -> bf16 conversion (weights)
// ---------------------------------------------------------------------------
__global__ void k_f32_to_bf16(const float* __restrict__ src,
                              unsigned short* __restrict__ dst, int n) {
    int i = blockIdx.x * blockDim.x + threadIdx.x;
    if (i < n) dst[i] = f2bf(src[i]);
}

// ---------------------------------------------------------------------------
// Kernel: GroupNorm statistics. Group g of batch b covers channels
// [g*64, g*64+64) which are contiguous in x[b,c,hw] -> contiguous 1 MB chunk.
// ---------------------------------------------------------------------------
__global__ __launch_bounds__(256) void k_gn_stats(const float* __restrict__ x,
                                                  float* __restrict__ stats) {
    const int bg = blockIdx.x;                   // b*8 + g, 32 blocks
    const float4* xv = (const float4*)(x + (size_t)bg * (64 * HW));
    float s = 0.f, ss = 0.f;
    for (int i = threadIdx.x; i < (64 * HW) / 4; i += 256) {
        float4 t = xv[i];
        s  += t.x + t.y + t.z + t.w;
        ss += t.x * t.x + t.y * t.y + t.z * t.z + t.w * t.w;
    }
    __shared__ float rs[256], rss[256];
    rs[threadIdx.x] = s; rss[threadIdx.x] = ss;
    __syncthreads();
    for (int st = 128; st > 0; st >>= 1) {
        if (threadIdx.x < st) {
            rs[threadIdx.x]  += rs[threadIdx.x + st];
            rss[threadIdx.x] += rss[threadIdx.x + st];
        }
        __syncthreads();
    }
    if (threadIdx.x == 0) {
        const float inv_n = 1.0f / (64.0f * 4096.0f);
        float mean = rs[0] * inv_n;
        float var  = rss[0] * inv_n - mean * mean;
        stats[bg * 2 + 0] = mean;
        stats[bg * 2 + 1] = rsqrtf(var + 1e-5f);
    }
}

// ---------------------------------------------------------------------------
// Kernel: GroupNorm apply. Writes xn fp32 (residual, [b,c,hw] into d_out) and
// transposed bf16 xnT[b,hw,c] (GEMM A-matrix layout; lives in L2).
// ---------------------------------------------------------------------------
__global__ __launch_bounds__(256) void k_gn_apply(
    const float* __restrict__ x, const float* __restrict__ stats,
    const float* __restrict__ gamma, const float* __restrict__ beta,
    float* __restrict__ xn_out, unsigned short* __restrict__ xnT) {
    int i = blockIdx.x * 256 + threadIdx.x;      // 8,388,608 threads
    int p = i & (HW - 1);
    int c = (i >> 12) & (C_DIM - 1);
    int b = i >> 21;
    int bg = b * 8 + (c >> 6);
    float mean = stats[bg * 2 + 0];
    float inv  = stats[bg * 2 + 1];
    float v = (x[i] - mean) * inv * gamma[c] + beta[c];
    xn_out[i] = v;
    xnT[(((size_t)((b << 12) | p)) << 9) + c] = f2bf(v);
}

// ---------------------------------------------------------------------------
// Kernel: WMMA GEMM  D[M=16384, N=512] = A[M,512](bf16) * W[512,512](bf16)^T + bias
// (W stored row-major [N,K] == the conv weight itself, which IS B^T.)
// Block tile 128x128, BK=32, 256 threads = 8 waves in 4x2 grid of 32x64 tiles.
// Double-buffered LDS staging via global_load_async_to_lds_b128 (ASYNCcnt).
// mode 0: bf16 out, row-major [m, n]              (q, k)
// mode 1: bf16 out, transposed [b, n, p]          (v)
// mode 2: fp32 accumulate into out [b, n, p]      (proj + residual)
// ---------------------------------------------------------------------------
__global__ __launch_bounds__(256) void k_gemm(
    const unsigned short* __restrict__ A, const unsigned short* __restrict__ W,
    const float* __restrict__ bias, void* __restrict__ outv, int mode) {
    __shared__ unsigned short As[2][128 * 32];
    __shared__ unsigned short Ws[2][128 * 32];

    const int tid = threadIdx.x;
    const int mBase = blockIdx.x * 128;
    const int nBase = blockIdx.y * 128;
    const int wave = tid >> 5, lane = tid & 31;
    const int wm = wave & 3, wn = wave >> 2;     // 4 x 2 wave grid
    const int laneN = lane & 15;
    const int laneHi = (lane & 16) ? 8 : 0;

    v8f acc[2][4];
#pragma unroll
    for (int i = 0; i < 2; ++i)
#pragma unroll
        for (int j = 0; j < 4; ++j) acc[i][j] = vzero8();

    // stage one 128x32 A tile + 128x32 W tile into buffer `buf` (async)
    auto stage = [&](int kt, int buf) {
        const int kBase = kt * 32;
#pragma unroll
        for (int c = tid; c < 512; c += 256) {   // 16B chunks; 2+2 per thread
            int row = c >> 2, col = (c & 3) << 3;
            async_ld16((unsigned)(size_t)&As[buf][row * 32 + col],
                       &A[(size_t)(mBase + row) * C_DIM + kBase + col]);
            async_ld16((unsigned)(size_t)&Ws[buf][row * 32 + col],
                       &W[(size_t)(nBase + row) * C_DIM + kBase + col]);
        }
    };

    stage(0, 0);
    for (int kt = 0; kt < 16; ++kt) {
        const int buf = kt & 1;
        if (kt + 1 < 16) {
            stage(kt + 1, buf ^ 1);   // prefetch next tile into other buffer
            WAIT_ASYNC(4);            // in-order: current tile's loads done
        } else {
            WAIT_ASYNC(0);
        }
        __syncthreads();

        v16bf a0 = ldfrag(As[buf], wm * 32 + 0, 32, 0);
        v16bf a1 = ldfrag(As[buf], wm * 32 + 16, 32, 0);
#pragma unroll
        for (int nf = 0; nf < 4; ++nf) {
            v16bf bfr = ldfrag(Ws[buf], wn * 64 + nf * 16, 32, 0);
            acc[0][nf] = __builtin_amdgcn_wmma_f32_16x16x32_bf16(
                false, a0, false, bfr, (short)0, acc[0][nf], false, false);
            acc[1][nf] = __builtin_amdgcn_wmma_f32_16x16x32_bf16(
                false, a1, false, bfr, (short)0, acc[1][nf], false, false);
        }
        __syncthreads();   // all waves done reading before buffer is re-filled
    }

    // Epilogue
#pragma unroll
    for (int mf = 0; mf < 2; ++mf) {
#pragma unroll
        for (int nf = 0; nf < 4; ++nf) {
            const int n = nBase + wn * 64 + nf * 16 + laneN;
            const float bv = bias[n];
#pragma unroll
            for (int e = 0; e < 8; ++e) {
                const int m = mBase + wm * 32 + mf * 16 + laneHi + e;
                const float val = acc[mf][nf][e] + bv;
                if (mode == 0) {
                    ((unsigned short*)outv)[(size_t)m * C_DIM + n] = f2bf(val);
                } else {
                    const int b = m >> 12, p = m & (HW - 1);
                    const size_t idx = (((size_t)(b * C_DIM + n)) << 12) + p;
                    if (mode == 1) ((unsigned short*)outv)[idx] = f2bf(val);
                    else           ((float*)outv)[idx] += val;   // residual add
                }
            }
        }
    }
}

// ---------------------------------------------------------------------------
// Kernel: flash attention. One block = 32 query rows of one batch.
// Loop 64-key tiles: S = Q K^T (WMMA, K=512), online softmax, O += P V (WMMA).
// Q staged once; K/V tiles double-buffered, all via async LDS copies.
// LDS: 32KB Q + 2*(64+64)KB K/V + ~13KB softmax state ~ 301 KB (< 320 KB WGP).
// Softmax parallelized 8 segments x 32 rows; running denominator kept in a
// private register per (row,segment) and reduced once at the end.
// ---------------------------------------------------------------------------
__global__ __launch_bounds__(256) void k_attn(
    const unsigned short* __restrict__ q, const unsigned short* __restrict__ k,
    const unsigned short* __restrict__ v, unsigned short* __restrict__ o) {
    __shared__ unsigned short Qs[32 * 512];      // [query i][channel c]
    __shared__ unsigned short Ks[2][64 * 512];   // [key j][channel c]
    __shared__ unsigned short Vs[2][512 * 64];   // [channel c][key j]
    __shared__ float          Ss[32 * 64];       // raw scaled scores
    __shared__ unsigned short Ps[32 * 64];       // exp probabilities (bf16)
    __shared__ float          Sm[256];           // per-(row,seg) reduce scratch
    __shared__ float mrow[32], lrow[32], arow[32];

    const int tid = threadIdx.x;
    const int wave = tid >> 5, lane = tid & 31;
    const int laneN = lane & 15;
    const int laneHi = (lane & 16) ? 8 : 0;
    const int qbase = blockIdx.x * 32;
    const int batch = blockIdx.y;
    const unsigned short* qblk = q + ((size_t)(batch * HW + qbase)) * C_DIM;

    if (tid < 32) mrow[tid] = -3.0e38f;
    v8f oacc[2][4];
#pragma unroll
    for (int i = 0; i < 2; ++i)
#pragma unroll
        for (int j = 0; j < 4; ++j) oacc[i][j] = vzero8();

    // stage the 32x512 Q block once (8 async chunks per thread)
#pragma unroll
    for (int cc = tid; cc < 2048; cc += 256) {
        int row = cc >> 6, col = (cc & 63) << 3;
        async_ld16((unsigned)(size_t)&Qs[(row << 9) + col],
                   &qblk[((size_t)row << 9) + col]);
    }

    // async stage of one K (64x512) + V ([512][64]) tile; 32 chunks per thread
    auto stage = [&](int jt, int buf) {
        const int jbase = jt * 64;
#pragma unroll
        for (int cc = tid; cc < 4096; cc += 256) {
            int row = cc >> 6, col = (cc & 63) << 3;
            async_ld16((unsigned)(size_t)&Ks[buf][(row << 9) + col],
                       &k[(((size_t)(batch * HW + jbase + row)) << 9) + col]);
        }
#pragma unroll
        for (int cc = tid; cc < 4096; cc += 256) {
            int ch = cc >> 3, col = (cc & 7) << 3;
            async_ld16((unsigned)(size_t)&Vs[buf][(ch << 6) + col],
                       &v[(((size_t)(batch * C_DIM + ch)) << 12) + jbase + col]);
        }
    };

    const int smf = wave >> 2;        // this wave's S fragment (row, col)
    const int snf = wave & 3;
    const int chBase = wave << 6;     // this wave's 64 output channels
    const int srow = tid >> 3;        // softmax: row 0..31
    const int sseg = tid & 7;         // softmax: segment 0..7 (8 cols each)
    float lseg = 0.0f;                // private running denominator share
    const float scale = 0.0441941738241592f;   // 1/sqrt(512)

    stage(0, 0);
    for (int jt = 0; jt < 64; ++jt) {
        const int buf = jt & 1;
        if (jt + 1 < 64) {
            stage(jt + 1, buf ^ 1);   // prefetch next K/V tile (overlaps compute)
            WAIT_ASYNC(32);           // in-order: Q + tile jt are resident
        } else {
            WAIT_ASYNC(0);
        }
        __syncthreads();

        // --- S = Q K^T : each wave one 16x16 fragment of the 32x64 tile ---
        v8f s = vzero8();
#pragma unroll
        for (int ks = 0; ks < 16; ++ks) {
            v16bf af  = ldfrag(Qs,      smf * 16, C_DIM, ks * 32);
            v16bf bfr = ldfrag(Ks[buf], snf * 16, C_DIM, ks * 32);
            s = __builtin_amdgcn_wmma_f32_16x16x32_bf16(
                false, af, false, bfr, (short)0, s, false, false);
        }
        {
            const int r0 = smf * 16 + laneHi, c0 = snf * 16 + laneN;
#pragma unroll
            for (int e = 0; e < 8; ++e) Ss[(r0 + e) * 64 + c0] = s[e] * scale;
        }
        __syncthreads();

        // --- online softmax, 8 threads per row ---
        const float* mySs = &Ss[srow * 64 + sseg * 8];
        {   // phase A: per-segment max
            float mx = mySs[0];
#pragma unroll
            for (int j = 1; j < 8; ++j) mx = fmaxf(mx, mySs[j]);
            Sm[tid] = mx;
        }
        __syncthreads();
        if (tid < 32) {   // phase B: cross-segment max, alpha
            float mo = mrow[tid], mx = mo;
#pragma unroll
            for (int s2 = 0; s2 < 8; ++s2) mx = fmaxf(mx, Sm[tid * 8 + s2]);
            arow[tid] = __expf(mo - mx);
            mrow[tid] = mx;
        }
        __syncthreads();
        {   // phase C: exps, Ps, private denominator update
            const float mx = mrow[srow];
            const float al = arow[srow];
            float ps = 0.0f;
#pragma unroll
            for (int j = 0; j < 8; ++j) {
                float pv = __expf(mySs[j] - mx);
                ps += pv;
                Ps[srow * 64 + sseg * 8 + j] = f2bf(pv);
            }
            lseg = lseg * al + ps;
        }
        __syncthreads();

        // --- O = diag(alpha) O + P V  (each wave owns 32x64 of O) ---
#pragma unroll
        for (int mf2 = 0; mf2 < 2; ++mf2)
#pragma unroll
            for (int nf2 = 0; nf2 < 4; ++nf2) {
                v8f t = oacc[mf2][nf2];
#pragma unroll
                for (int e = 0; e < 8; ++e) t[e] *= arow[mf2 * 16 + laneHi + e];
                oacc[mf2][nf2] = t;
            }
#pragma unroll
        for (int ks2 = 0; ks2 < 2; ++ks2) {
            v16bf pa0 = ldfrag(Ps, 0,  64, ks2 * 32);
            v16bf pa1 = ldfrag(Ps, 16, 64, ks2 * 32);
#pragma unroll
            for (int nf2 = 0; nf2 < 4; ++nf2) {
                v16bf vb = ldfrag(Vs[buf], chBase + nf2 * 16, 64, ks2 * 32);
                oacc[0][nf2] = __builtin_amdgcn_wmma_f32_16x16x32_bf16(
                    false, pa0, false, vb, (short)0, oacc[0][nf2], false, false);
                oacc[1][nf2] = __builtin_amdgcn_wmma_f32_16x16x32_bf16(
                    false, pa1, false, vb, (short)0, oacc[1][nf2], false, false);
            }
        }
        __syncthreads();   // tile fully consumed; safe to refill this buffer
    }

    // --- reduce denominator across segments ---
    Sm[tid] = lseg;
    __syncthreads();
    if (tid < 32) {
        float l = 0.0f;
#pragma unroll
        for (int s2 = 0; s2 < 8; ++s2) l += Sm[tid * 8 + s2];
        lrow[tid] = l;
    }
    __syncthreads();

    // --- normalize by running sum, store O^T[b, hw, c] bf16 ---
#pragma unroll
    for (int mf2 = 0; mf2 < 2; ++mf2)
#pragma unroll
        for (int nf2 = 0; nf2 < 4; ++nf2) {
#pragma unroll
            for (int e = 0; e < 8; ++e) {
                const int r  = mf2 * 16 + laneHi + e;
                const int ch = chBase + nf2 * 16 + laneN;
                const float val = oacc[mf2][nf2][e] / lrow[r];
                o[(((size_t)(batch * HW + qbase + r)) << 9) + ch] = f2bf(val);
            }
        }
}

// ---------------------------------------------------------------------------
// Host-side launcher
// ---------------------------------------------------------------------------
extern "C" void kernel_launch(void* const* d_in, const int* in_sizes, int n_in,
                              void* d_out, int out_size, void* d_ws, size_t ws_size,
                              hipStream_t stream) {
    const float* x     = (const float*)d_in[0];
    const float* gamma = (const float*)d_in[1];
    const float* beta  = (const float*)d_in[2];
    const float* wq = (const float*)d_in[3];
    const float* bq = (const float*)d_in[4];
    const float* wk = (const float*)d_in[5];
    const float* bk = (const float*)d_in[6];
    const float* wv = (const float*)d_in[7];
    const float* bv = (const float*)d_in[8];
    const float* wp = (const float*)d_in[9];
    const float* bp = (const float*)d_in[10];
    float* out = (float*)d_out;

    // Workspace layout (all 1 KB aligned)
    char* ws = (char*)d_ws;
    float* stats = (float*)ws;                                  // 256 B
    unsigned short* wqb = (unsigned short*)(ws + 1024);
    unsigned short* wkb = wqb + 262144;
    unsigned short* wvb = wkb + 262144;
    unsigned short* wpb = wvb + 262144;
    unsigned short* xnT = wpb + 262144;       // [b, hw, c]  16 MB
    unsigned short* qT  = xnT + 8388608;      // [b, hw, c]  16 MB
    unsigned short* kT  = qT  + 8388608;      // [b, hw, c]  16 MB
    unsigned short* vB  = kT  + 8388608;      // [b, c, hw]  16 MB
    unsigned short* oT  = vB  + 8388608;      // [b, hw, c]  16 MB

    // 1) weights -> bf16
    k_f32_to_bf16<<<1024, 256, 0, stream>>>(wq, wqb, 262144);
    k_f32_to_bf16<<<1024, 256, 0, stream>>>(wk, wkb, 262144);
    k_f32_to_bf16<<<1024, 256, 0, stream>>>(wv, wvb, 262144);
    k_f32_to_bf16<<<1024, 256, 0, stream>>>(wp, wpb, 262144);

    // 2) GroupNorm
    k_gn_stats<<<32, 256, 0, stream>>>(x, stats);
    k_gn_apply<<<32768, 256, 0, stream>>>(x, stats, gamma, beta, out, xnT);

    // 3) Q, K, V projections (WMMA GEMMs)
    dim3 gg(M_TOT / 128, C_DIM / 128);   // (128, 4)
    k_gemm<<<gg, 256, 0, stream>>>(xnT, wqb, bq, qT, 0);
    k_gemm<<<gg, 256, 0, stream>>>(xnT, wkb, bk, kT, 0);
    k_gemm<<<gg, 256, 0, stream>>>(xnT, wvb, bv, vB, 1);

    // 4) flash attention
    dim3 ga(HW / 32, BATCH);             // (128, 4)
    k_attn<<<ga, 256, 0, stream>>>(qT, kT, vB, oT);

    // 5) projection + residual accumulate into d_out (which holds xn)
    k_gemm<<<gg, 256, 0, stream>>>(oT, wpb, bp, out, 2);
}
